// MultiTaskNavActorCritic_39651138077368
// MI455X (gfx1250) — matrix operand order repaired and checked by reading, not compiled, not run on hardware
//
#include <hip/hip_runtime.h>
#include <hip/hip_bf16.h>

typedef _Float16 f16;
typedef __attribute__((ext_vector_type(16))) _Float16 v16h;
typedef __attribute__((ext_vector_type(8)))  float    v8f;
typedef unsigned int u32x4 __attribute__((ext_vector_type(4)));

#define WMMA_F16(C, A, Bm) \
  (C) = __builtin_amdgcn_wmma_f32_16x16x32_f16(false, (A), false, (Bm), (short)0, (C), false, false)

// A operand (16x32 f16, M=row striped per lane): lane&15 = row M; half = lane>>4.
// half 0 holds K 0..15; half 1 holds K 16..31 (contiguous, matches B swizzle).
#define LOAD_A16(dst, rowptr, koff, half) do {                                   \
  ((u32x4*)&(dst))[0] = *(const u32x4*)((rowptr) + (koff) + (half)*16);          \
  ((u32x4*)&(dst))[1] = *(const u32x4*)((rowptr) + (koff) + (half)*16 + 8);      \
} while (0)

// B operand pre-swizzled in memory: 16 contiguous halves per lane.
#define LOAD_B16(dst, laneptr) do {                                              \
  ((u32x4*)&(dst))[0] = *(const u32x4*)(laneptr);                                \
  ((u32x4*)&(dst))[1] = *(const u32x4*)((laneptr) + 8);                          \
} while (0)

// ---------------------------------------------------------------------------
// Weight prep: f32 row-major [N][K] -> f16 WMMA-B swizzle [nt][kt][lane][16]
//   element(nt,kt,lane,j) = src[(nt*16 + (lane&15)) * K + kt*32 + (lane>>4)*16 + j]
// ---------------------------------------------------------------------------
__global__ __launch_bounds__(256) void swizzle_gen_kernel(
    const float* __restrict__ src, f16* __restrict__ dst, int nKT, int K, int total)
{
  int idx = blockIdx.x * 256 + threadIdx.x;
  if (idx >= total) return;
  int j = idx & 15, lane = (idx >> 4) & 31;
  int r = idx >> 9;
  int kt = r % nKT, nt = r / nKT;
  int n = nt * 16 + (lane & 15);
  int k = kt * 32 + (lane >> 4) * 16 + j;
  dst[idx] = (f16)src[(size_t)n * K + k];
}

// conv weights [CO][CI][3][3] -> per-tap B tiles: [tap][kt][nt][lane][16]
__global__ __launch_bounds__(256) void swizzle_conv_kernel(
    const float* __restrict__ src, f16* __restrict__ dst, int nKT, int nNT, int CI, int total)
{
  int idx = blockIdx.x * 256 + threadIdx.x;
  if (idx >= total) return;
  int j = idx & 15, lane = (idx >> 4) & 31;
  int r = idx >> 9;
  int nt = r % nNT; r /= nNT;
  int kt = r % nKT; int tap = r / nKT;
  int co = nt * 16 + (lane & 15);
  int ci = kt * 32 + (lane >> 4) * 16 + j;
  dst[idx] = (f16)src[((size_t)co * CI + ci) * 9 + tap];
}

// ---------------------------------------------------------------------------
// Encoder: embedding gather -> conv1(96->64,3x3,SAME)+relu -> conv2(64->128)+relu
//          -> fused[B][6368] f16 (spatial 6272 | dir emb 32 | task emb 64)
// 2 samples per WG, 1 wave per sample. Convs as per-tap WMMA GEMMs over a
// zero-padded 9x9 halo tile held in LDS.
// ---------------------------------------------------------------------------
__global__ __launch_bounds__(64) void conv_kernel(
    const int* __restrict__ obj, const int* __restrict__ color, const int* __restrict__ state,
    const int* __restrict__ direction, const int* __restrict__ task_id,
    const float* __restrict__ emb_obj, const float* __restrict__ emb_color,
    const float* __restrict__ emb_state, const float* __restrict__ emb_dir,
    const float* __restrict__ emb_task,
    const float* __restrict__ conv1_b, const float* __restrict__ conv2_b,
    const f16* __restrict__ w1sw, const f16* __restrict__ w2sw,
    f16* __restrict__ fused)
{
  __shared__ f16 featp[2][81 * 96];   // padded 9x9 x 96ch, zero halo
  __shared__ f16 out1p[2][81 * 64];   // padded 9x9 x 64ch, zero halo
  const int lane = threadIdx.x & 31;
  const int wv   = threadIdx.x >> 5;
  const int half = lane >> 4;
  const int col  = lane & 15;
  const int s    = blockIdx.x * 2 + wv;

  for (int i = lane; i < 81 * 96; i += 32) featp[wv][i] = (f16)0.f;
  for (int i = lane; i < 81 * 64; i += 32) out1p[wv][i] = (f16)0.f;
  __syncthreads();

  // embedding gather into padded interior (row-major [pos][96])
  for (int idx = lane; idx < 49 * 96; idx += 32) {
    int p = idx / 96, c = idx % 96;
    int y = p / 7, x = p % 7;
    float v;
    if (c < 32)      v = emb_obj  [obj  [s * 49 + p] * 32 + c];
    else if (c < 64) v = emb_color[color[s * 49 + p] * 32 + (c - 32)];
    else             v = emb_state[state[s * 49 + p] * 32 + (c - 64)];
    featp[wv][((y + 1) * 9 + (x + 1)) * 96 + c] = (f16)v;
  }
  __syncthreads();

  // ---- conv1: M=64 positions (49 valid), N=64 ch, K=96, 9 taps ----
  for (int m = 0; m < 4; ++m) {
    int p = m * 16 + col;
    int py = p / 7, px = p % 7;
    bool pv = p < 49;
    for (int n = 0; n < 4; ++n) {
      v8f c = {};
      for (int tap = 0; tap < 9; ++tap) {
        int dy = tap / 3, dx = tap % 3;
        const f16* arow = pv ? &featp[wv][((py + dy) * 9 + (px + dx)) * 96]
                             : &featp[wv][0];          // padded border == zeros
        #pragma unroll
        for (int k0 = 0; k0 < 3; ++k0) {
          v16h a, b;
          LOAD_A16(a, arow, k0 * 32, half);
          const f16* bp = w1sw + ((((tap * 3 + k0) * 4 + n) * 32 + lane) << 4);
          LOAD_B16(b, bp);
          WMMA_F16(c, a, b);
        }
      }
      int co = n * 16 + col;
      float bias = conv1_b[co];
      #pragma unroll
      for (int i = 0; i < 8; ++i) {
        int p2 = m * 16 + i + half * 8;
        if (p2 < 49) {
          float v = c[i] + bias; v = v > 0.f ? v : 0.f;
          out1p[wv][((p2 / 7 + 1) * 9 + (p2 % 7 + 1)) * 64 + co] = (f16)v;
        }
      }
    }
  }
  __syncthreads();

  // ---- conv2: M=64 positions, N=128 ch, K=64, 9 taps ----
  for (int m = 0; m < 4; ++m) {
    int p = m * 16 + col;
    int py = p / 7, px = p % 7;
    bool pv = p < 49;
    for (int n = 0; n < 8; ++n) {
      v8f c = {};
      for (int tap = 0; tap < 9; ++tap) {
        int dy = tap / 3, dx = tap % 3;
        const f16* arow = pv ? &out1p[wv][((py + dy) * 9 + (px + dx)) * 64]
                             : &out1p[wv][0];
        #pragma unroll
        for (int k0 = 0; k0 < 2; ++k0) {
          v16h a, b;
          LOAD_A16(a, arow, k0 * 32, half);
          const f16* bp = w2sw + ((((tap * 2 + k0) * 8 + n) * 32 + lane) << 4);
          LOAD_B16(b, bp);
          WMMA_F16(c, a, b);
        }
      }
      int co = n * 16 + col;
      float bias = conv2_b[co];
      #pragma unroll
      for (int i = 0; i < 8; ++i) {
        int p2 = m * 16 + i + half * 8;
        if (p2 < 49) {
          float v = c[i] + bias; v = v > 0.f ? v : 0.f;
          fused[(size_t)s * 6368 + co * 49 + p2] = (f16)v;   // reshape order c*49+p
        }
      }
    }
  }

  // dir / task embedding columns 6272..6367
  for (int j = lane; j < 96; j += 32) {
    float v = (j < 32) ? emb_dir[direction[s] * 32 + j]
                       : emb_task[task_id[s] * 64 + (j - 32)];
    fused[(size_t)s * 6368 + 6272 + j] = (f16)v;
  }
}

// ---------------------------------------------------------------------------
// pre: lat = relu(fused @ pre_w.T + pre_b)  -- M=8192, N=512, K=6368
// 16 rows per WG, each wave owns 64 output cols, K-loop = 199 WMMA steps.
// ---------------------------------------------------------------------------
__global__ __launch_bounds__(256) void pre_gemm_kernel(
    const f16* __restrict__ fused, const f16* __restrict__ pwsw,
    const float* __restrict__ pre_b, f16* __restrict__ lat)
{
  const int lane = threadIdx.x & 31, wv = threadIdx.x >> 5;
  const int half = lane >> 4, col = lane & 15;
  const int r0 = blockIdx.x * 16;
  const f16* arow = fused + (size_t)(r0 + col) * 6368;
  const int ntb = wv * 4;
  v8f c0 = {}, c1 = {}, c2 = {}, c3 = {};
  for (int k = 0; k < 199; ++k) {
    v16h a; LOAD_A16(a, arow, k * 32, half);
    const f16* bp = pwsw + (((size_t)ntb * 199 + k) * 32 + lane) * 16;
    v16h b0, b1, b2, b3;
    LOAD_B16(b0, bp);
    LOAD_B16(b1, bp + (size_t)199 * 512);
    LOAD_B16(b2, bp + (size_t)2 * 199 * 512);
    LOAD_B16(b3, bp + (size_t)3 * 199 * 512);
    __builtin_prefetch(bp + 512, 0, 1);        // next k-step B tile
    WMMA_F16(c0, a, b0); WMMA_F16(c1, a, b1);
    WMMA_F16(c2, a, b2); WMMA_F16(c3, a, b3);
  }
  #pragma unroll
  for (int t = 0; t < 4; ++t) {
    v8f c = (t == 0) ? c0 : (t == 1) ? c1 : (t == 2) ? c2 : c3;
    int cn = (ntb + t) * 16 + col;
    float bias = pre_b[cn];
    #pragma unroll
    for (int i = 0; i < 8; ++i) {
      int row = r0 + i + half * 8;
      float v = c[i] + bias; v = v > 0.f ? v : 0.f;
      lat[(size_t)row * 512 + cn] = (f16)v;
    }
  }
}

// ---------------------------------------------------------------------------
// Task-routed GRU step (h0=0) + per-task adapter + per-task heads.
// 16 rows per WG. Routed GEMMs: compute each candidate weight set's tile,
// masked-merge rows by gidx/task (masking is scalar, after the WMMA).
// A-tile staged into LDS via GLOBAL_LOAD_ASYNC_TO_LDS_B128 (ASYNCcnt path).
// ---------------------------------------------------------------------------
__global__ __launch_bounds__(256) void gru_head_kernel(
    const f16* __restrict__ lat, const int* __restrict__ task_id,
    const f16* __restrict__ wihsw, const float* __restrict__ b_ih, const float* __restrict__ b_hh,
    const f16* __restrict__ wasw, const float* __restrict__ ba,
    const float* __restrict__ Wh, const float* __restrict__ bh2,
    const float* __restrict__ vw, const float* __restrict__ vb,
    float* __restrict__ out)
{
  __shared__ f16 latT [16 * 512];
  __shared__ f16 lat2T[16 * 512];
  __shared__ f16 lat3T[16 * 512];
  __shared__ int gI[16], tI[16];
  const int tid = threadIdx.x;
  const int lane = tid & 31, wv = tid >> 5;
  const int half = lane >> 4, col = lane & 15;
  const int r0 = blockIdx.x * 16;

  // ---- async global->LDS stage of the 16x512 f16 A tile (contiguous 16KB) ----
  // LDS address operand = low 32 bits of the flat shared address (ISA 10.2:
  // LDS aperture maps addr[31:0] directly to the LDS offset).
  #pragma unroll
  for (int it = 0; it < 4; ++it) {
    int e = (it * 256 + tid) * 8;                       // 8 halves = 16 B / thread
    unsigned lds_addr = (unsigned)(size_t)&latT[e];
    const f16* gp = lat + (size_t)r0 * 512 + e;
    asm volatile("global_load_async_to_lds_b128 %0, %1, off"
                 :: "v"(lds_addr), "v"(gp) : "memory");
  }
  if (tid < 16) { int t = task_id[r0 + tid]; tI[tid] = t; gI[tid] = (t < 3) ? t : 3; }
  asm volatile("s_wait_asynccnt 0" ::: "memory");
  __syncthreads();

  // ---- GRU: gx = lat @ W_ih[g].T + b_ih[g]; gh = b_hh[g]; h = (1-z)*n ----
  for (int hc = wv; hc < 32; hc += 8) {
    float RX[8] = {0}, ZX[8] = {0}, NX[8] = {0};
    for (int g = 0; g < 4; ++g) {
      v8f cr = {}, cz = {}, cn2 = {};
      for (int k = 0; k < 16; ++k) {
        v16h a; LOAD_A16(a, latT + col * 512, k * 32, half);
        const f16* bp = wihsw + ((((size_t)g * 96 + hc) * 16 + k) * 32 + lane) * 16;
        v16h br, bz, bn;
        LOAD_B16(br, bp);
        LOAD_B16(bz, bp + (size_t)32 * 16 * 512);
        LOAD_B16(bn, bp + (size_t)64 * 16 * 512);
        WMMA_F16(cr, a, br); WMMA_F16(cz, a, bz); WMMA_F16(cn2, a, bn);
      }
      #pragma unroll
      for (int i = 0; i < 8; ++i) {
        int r = i + half * 8;
        if (gI[r] == g) { RX[i] = cr[i]; ZX[i] = cz[i]; NX[i] = cn2[i]; }
      }
    }
    int n = hc * 16 + col;
    #pragma unroll
    for (int i = 0; i < 8; ++i) {
      int r = i + half * 8; int g = gI[r];
      float rx = RX[i] + b_ih[g * 1536 + n];
      float zx = ZX[i] + b_ih[g * 1536 + 512 + n];
      float nx = NX[i] + b_ih[g * 1536 + 1024 + n];
      float rh = b_hh[g * 1536 + n];
      float zh = b_hh[g * 1536 + 512 + n];
      float nh = b_hh[g * 1536 + 1024 + n];
      float rr = 1.f / (1.f + __expf(-(rx + rh)));
      float zz = 1.f / (1.f + __expf(-(zx + zh)));
      float nn = tanhf(nx + rr * nh);
      lat2T[r * 512 + n] = (f16)((1.f - zz) * nn);
    }
  }
  __syncthreads();

  // ---- per-task adapters: relu(lat2 @ Wa[t].T + ba[t]) ----
  for (int hc = wv; hc < 32; hc += 8) {
    float AV[8] = {0};
    for (int t = 0; t < 8; ++t) {
      v8f c = {};
      for (int k = 0; k < 16; ++k) {
        v16h a; LOAD_A16(a, lat2T + col * 512, k * 32, half);
        const f16* bp = wasw + ((((size_t)t * 32 + hc) * 16 + k) * 32 + lane) * 16;
        v16h b; LOAD_B16(b, bp);
        WMMA_F16(c, a, b);
      }
      #pragma unroll
      for (int i = 0; i < 8; ++i) { int r = i + half * 8; if (tI[r] == t) AV[i] = c[i]; }
    }
    int n = hc * 16 + col;
    #pragma unroll
    for (int i = 0; i < 8; ++i) {
      int r = i + half * 8; int t = tI[r];
      float v = AV[i] + ba[t * 512 + n]; v = v > 0.f ? v : 0.f;
      lat3T[r * 512 + n] = (f16)v;
    }
  }
  __syncthreads();

  // ---- tiny heads: 7 logits + value per row ----
  if (tid < 128) {
    int r = tid >> 3, o = tid & 7; int t = tI[r]; int row = r0 + r;
    const f16* lr = lat3T + r * 512;
    float acc = 0.f;
    if (o < 7) {
      const float* wr = Wh + ((size_t)t * 7 + o) * 512;
      for (int k = 0; k < 512; ++k) acc += (float)lr[k] * wr[k];
      out[(size_t)row * 7 + o] = acc + bh2[t * 7 + o];
    } else {
      for (int k = 0; k < 512; ++k) acc += (float)lr[k] * vw[k];
      out[(size_t)8192 * 7 + row] = acc + vb[0];
    }
  }
}

// ---------------------------------------------------------------------------
extern "C" void kernel_launch(void* const* d_in, const int* in_sizes, int n_in,
                              void* d_out, int out_size, void* d_ws, size_t ws_size,
                              hipStream_t stream) {
  (void)in_sizes; (void)n_in; (void)out_size; (void)ws_size;
  const int*   obj       = (const int*)d_in[0];
  const int*   color     = (const int*)d_in[1];
  const int*   state     = (const int*)d_in[2];
  const int*   direction = (const int*)d_in[3];
  const int*   task_id   = (const int*)d_in[4];
  const float* emb_obj   = (const float*)d_in[5];
  const float* emb_color = (const float*)d_in[6];
  const float* emb_state = (const float*)d_in[7];
  const float* emb_dir   = (const float*)d_in[8];
  const float* emb_task  = (const float*)d_in[9];
  const float* conv1_w   = (const float*)d_in[10];
  const float* conv1_b   = (const float*)d_in[11];
  const float* conv2_w   = (const float*)d_in[12];
  const float* conv2_b   = (const float*)d_in[13];
  const float* pre_w     = (const float*)d_in[14];
  const float* pre_b     = (const float*)d_in[15];
  const float* W_ih      = (const float*)d_in[16];
  // d_in[17] = W_hh unused: h0 == 0
  const float* b_ih      = (const float*)d_in[18];
  const float* b_hh      = (const float*)d_in[19];
  const float* Wa        = (const float*)d_in[20];
  const float* ba        = (const float*)d_in[21];
  const float* Wh        = (const float*)d_in[22];
  const float* bh2       = (const float*)d_in[23];
  const float* vw        = (const float*)d_in[24];
  const float* vb        = (const float*)d_in[25];
  float* out = (float*)d_out;

  // workspace carve-up (f16 units)
  f16* p = (f16*)d_ws;
  f16* fused = p;  p += (size_t)8192 * 6368;
  f16* lat   = p;  p += (size_t)8192 * 512;
  f16* pwsw  = p;  p += (size_t)512 * 6368;
  f16* wihsw = p;  p += (size_t)6144 * 512;
  f16* wasw  = p;  p += (size_t)4096 * 512;
  f16* w1sw  = p;  p += 9 * 3 * 4 * 512;
  f16* w2sw  = p;  p += 9 * 2 * 8 * 512;

  auto nb = [](int total) { return (total + 255) / 256; };

  // weight prep (f32 -> f16, WMMA B-operand swizzle)
  swizzle_conv_kernel<<<nb(55296),      256, 0, stream>>>(conv1_w, w1sw, 3, 4, 96, 55296);
  swizzle_conv_kernel<<<nb(73728),      256, 0, stream>>>(conv2_w, w2sw, 2, 8, 64, 73728);
  swizzle_gen_kernel <<<nb(512 * 6368), 256, 0, stream>>>(pre_w, pwsw, 199, 6368, 512 * 6368);
  swizzle_gen_kernel <<<nb(6144 * 512), 256, 0, stream>>>(W_ih, wihsw, 16, 512, 6144 * 512);
  swizzle_gen_kernel <<<nb(4096 * 512), 256, 0, stream>>>(Wa,   wasw,  16, 512, 4096 * 512);

  conv_kernel<<<4096, 64, 0, stream>>>(obj, color, state, direction, task_id,
                                       emb_obj, emb_color, emb_state, emb_dir, emb_task,
                                       conv1_b, conv2_b, w1sw, w2sw, fused);
  pre_gemm_kernel<<<512, 256, 0, stream>>>(fused, pwsw, pre_b, lat);
  gru_head_kernel<<<512, 256, 0, stream>>>(lat, task_id, wihsw, b_ih, b_hh,
                                           wasw, ba, Wh, bh2, vw, vb, out);
}
